// Encoder_14061722927205
// MI455X (gfx1250) — matrix operand order repaired and checked by reading, not compiled, not run on hardware
//
#include <hip/hip_runtime.h>
#include <hip/hip_bf16.h>
#include <stdint.h>

typedef __bf16 bf16_t;
typedef __attribute__((ext_vector_type(16))) __bf16 v16bf;
typedef __attribute__((ext_vector_type(8)))  __bf16 v8bf;
typedef __attribute__((ext_vector_type(8)))  float  v8f;
typedef __attribute__((ext_vector_type(4)))  float  v4f;

#define B_  4
#define C_  512
#define T_  1024
#define H_  8
#define DK_ 64
#define F_  2048
#define L_  6
#define W_  4

#define TILE_M 128
#define TILE_N 64
#define TILE_K 32
#define LDSA_STRIDE 40   /* halfs per A row: 32 + 8 pad (80B, 16B aligned) */
#define LDSB_STRIDE 72   /* halfs per B row: 64 + 8 pad (144B, 16B aligned) */

// ---------------------------------------------------------------------------
// CDNA5 async global->LDS copy (ASYNCcnt-tracked DMA-style copy, no VGPR data).
// LDS aperture keeps the byte offset in addr[31:0], so truncating the generic
// pointer yields the VDST LDS address the instruction expects.
// ---------------------------------------------------------------------------
__device__ __forceinline__ void async_b128_to_lds(bf16_t* lds_ptr, const bf16_t* gptr) {
#if defined(__gfx1250__)
  const unsigned lds_off = (unsigned)(unsigned long long)lds_ptr;
  asm volatile("global_load_async_to_lds_b128 %0, %1, off"
               :: "v"(lds_off), "v"(gptr)
               : "memory");
#else
  *reinterpret_cast<v8bf*>(lds_ptr) = *reinterpret_cast<const v8bf*>(gptr);
#endif
}
__device__ __forceinline__ void wait_async() {
#if defined(__gfx1250__)
  asm volatile("s_wait_asynccnt 0x0" ::: "memory");
#endif
}

struct GemmParams {
  const bf16_t* A;
  const bf16_t* B;
  float*  outF;
  bf16_t* outH;
  const float* bias;    // indexed by m, nullable
  const float* maskN;   // per-n multiplier with batch stride, nullable
  int M, N, K;
  int lda, ldb, ldc;
  long long sAb, sAh, sBb, sBh, sDb, sDh, sMaskB;
  int Hdim;
  int ntap; long long tapStrideA; int tapShift0;
  float alpha;
  int relu, transOut;
};

// ---------------------------------------------------------------------------
// Batched WMMA GEMM: D = act(alpha*(A*B + bias)) * maskN
// A: bf16 [M,K] row-major (ld=lda), optionally 3 "tap" matrices (conv1d-K3).
// TRANSB=1: B given as [N,K] row-major (Bt rows direct; supports n-shift/taps)
// TRANSB=0: B given as [K,N] row-major
// 256 threads = 8 wave32; each wave owns a 32x32 subtile (2x2 WMMA tiles).
// Double-buffered LDS, async global->LDS staging overlapped with WMMA.
// ---------------------------------------------------------------------------
template <bool TRANSB>
__global__ __launch_bounds__(256) void k_gemm(GemmParams p) {
  __shared__ __align__(16) bf16_t sA[2][TILE_M * LDSA_STRIDE];
  __shared__ __align__(16) bf16_t sB[2][TILE_K * LDSB_STRIDE];
  const int tid   = threadIdx.x;
  const int wave  = tid >> 5;
  const int lane  = tid & 31;
  const int lrow  = lane & 15;
  const int lhalf = lane >> 4;
  const int z  = blockIdx.z;
  const int bb = z / p.Hdim;
  const int hh = z % p.Hdim;
  const bf16_t* A0 = p.A + (long long)bb * p.sAb + (long long)hh * p.sAh;
  const bf16_t* B0 = p.B + (long long)bb * p.sBb + (long long)hh * p.sBh;
  const int m0 = blockIdx.y * TILE_M;
  const int n0 = blockIdx.x * TILE_N;
  const int wm = (wave >> 1) * 32;   // 4 wave-rows cover 128
  const int wn = (wave & 1) * 32;    // 2 wave-cols cover 64
  const int kchunks = p.K >> 5;      // K chunks per tap
  const int total   = p.ntap * kchunks;

  // stage tile `chunk` (tap-major) into LDS buffer `buf`
  auto stage = [&](int chunk, int buf) {
    const int tap = chunk / kchunks;
    const int k0  = (chunk - tap * kchunks) << 5;
    const bf16_t* Atap = A0 + (long long)tap * p.tapStrideA;
    const int nshift = (p.ntap > 1) ? (p.tapShift0 + tap) : 0;
    bf16_t* sa = sA[buf];
    bf16_t* sb = sB[buf];
    // ---- A tile [128 rows][32 halfs]: 512 async b128 copies
#pragma unroll
    for (int it = 0; it < 2; ++it) {
      const int idx = tid + it * 256;
      const int r   = idx >> 2;
      const int c8  = (idx & 3) * 8;
      async_b128_to_lds(&sa[r * LDSA_STRIDE + c8],
                        Atap + (long long)(m0 + r) * p.lda + (k0 + c8));
    }
    // ---- B tile -> LDS [32 k-rows][64 n-cols]
    if (TRANSB) {
      const bool interior = (n0 + nshift >= 0) && (n0 + TILE_N - 1 + nshift < p.N);
      if (interior) {
        // branch-free: one b128 global load per thread, 8 strided LDS stores
        const int nn = tid >> 2;
        const int c8 = (tid & 3) * 8;
        const v8bf v = *reinterpret_cast<const v8bf*>(
            B0 + (long long)(n0 + nn + nshift) * p.ldb + (k0 + c8));
#pragma unroll
        for (int j = 0; j < 8; ++j) sb[(c8 + j) * LDSB_STRIDE + nn] = v[j];
      } else {
        // conv edge tiles only: scalar with bounds check (zero pad)
#pragma unroll
        for (int it = 0; it < 8; ++it) {
          const int idx = tid + it * 256;
          const int kk  = idx & 31;
          const int nn  = idx >> 5;
          const int nrow = n0 + nn + nshift;
          bf16_t val = (bf16_t)0.0f;
          if (nrow >= 0 && nrow < p.N)
            val = B0[(long long)nrow * p.ldb + (k0 + kk)];
          sb[kk * LDSB_STRIDE + nn] = val;
        }
      }
    } else {
      // B already [K,N]: 256 async b128 row copies
      const int kk = tid >> 3;
      const int c8 = (tid & 7) * 8;
      async_b128_to_lds(&sb[kk * LDSB_STRIDE + c8],
                        B0 + (long long)(k0 + kk) * p.ldb + (n0 + c8));
    }
  };

  v8f acc[2][2];
#pragma unroll
  for (int i = 0; i < 2; ++i)
#pragma unroll
    for (int j = 0; j < 2; ++j)
#pragma unroll
      for (int r = 0; r < 8; ++r) acc[i][j][r] = 0.0f;

  stage(0, 0);
  for (int c = 0; c < total; ++c) {
    wait_async();          // own copies for buffer c&1 done
    __syncthreads();       // everyone's copies done; prior reads of alt buf done
    if (c + 1 < total) stage(c + 1, (c + 1) & 1);   // prefetch next tile
    const bf16_t* sa = sA[c & 1];
    const bf16_t* sb = sB[c & 1];
    // ---- fragments (ISA 7.12.2 layouts) + 32 WMMAs per K-chunk (per wave: 4)
    v16bf af[2], bfr[2];
#pragma unroll
    for (int i = 0; i < 2; ++i) {
      const bf16_t* pa = &sa[(wm + i * 16 + lrow) * LDSA_STRIDE + lhalf * 8];
      v8bf a0 = *reinterpret_cast<const v8bf*>(pa);
      v8bf a1 = *reinterpret_cast<const v8bf*>(pa + 16);
      af[i] = __builtin_shufflevector(a0, a1, 0,1,2,3,4,5,6,7,8,9,10,11,12,13,14,15);
      const bf16_t* pb = &sb[(lhalf * 8) * LDSB_STRIDE + wn + i * 16 + lrow];
#pragma unroll
      for (int s = 0; s < 8; ++s) {
        bfr[i][s]     = pb[s * LDSB_STRIDE];
        bfr[i][s + 8] = pb[(s + 16) * LDSB_STRIDE];
      }
    }
#pragma unroll
    for (int i = 0; i < 2; ++i)
#pragma unroll
      for (int j = 0; j < 2; ++j)
        acc[i][j] = __builtin_amdgcn_wmma_f32_16x16x32_bf16(
            false, af[i], false, bfr[j], (short)0, acc[i][j], false, false);
  }

  // ---- epilogue
  float*  outF = p.outF ? p.outF + (long long)bb * p.sDb + (long long)hh * p.sDh : nullptr;
  bf16_t* outH = p.outH ? p.outH + (long long)bb * p.sDb + (long long)hh * p.sDh : nullptr;
#pragma unroll
  for (int i = 0; i < 2; ++i) {
#pragma unroll
    for (int j = 0; j < 2; ++j) {
      const int n = n0 + wn + j * 16 + lrow;
      const float mk = p.maskN ? p.maskN[(long long)bb * p.sMaskB + n] : 1.0f;
      const int mbase = m0 + wm + i * 16 + lhalf * 8;
      float v[8];
#pragma unroll
      for (int r = 0; r < 8; ++r) {
        float t = acc[i][j][r];
        if (p.bias) t += p.bias[mbase + r];
        t *= p.alpha;                       // reference scales (conv+bias)
        if (p.relu) t = t > 0.0f ? t : 0.0f;
        v[r] = t * mk;
      }
      if (p.transOut) {
        // contiguous in m per lane -> vector stores
        const long long off = (long long)n * p.ldc + mbase;
        if (outF) {
          v4f lo = {v[0], v[1], v[2], v[3]};
          v4f hi = {v[4], v[5], v[6], v[7]};
          *reinterpret_cast<v4f*>(outF + off)     = lo;
          *reinterpret_cast<v4f*>(outF + off + 4) = hi;
        }
        if (outH) {
          v8bf hv;
#pragma unroll
          for (int r = 0; r < 8; ++r) hv[r] = (bf16_t)v[r];
          *reinterpret_cast<v8bf*>(outH + off) = hv;
        }
      } else {
        // coalesced across lanes (n = lane%16 consecutive)
#pragma unroll
        for (int r = 0; r < 8; ++r) {
          const long long off = (long long)(mbase + r) * p.ldc + n;
          if (outF) outF[off] = v[r];
          if (outH) outH[off] = (bf16_t)v[r];
        }
      }
    }
  }
}

// ---------------------------------------------------------------------------
// Softmax with windowed rel-K bias and mask. One block per (t, b*H+h) row.
// rel bias nonzero only for e = s - t + W in [0, 2W] (zero-padded embedding).
// ---------------------------------------------------------------------------
__global__ __launch_bounds__(256) void k_softmax_rel(const float* scores, bf16_t* P,
                                                     const bf16_t* qbuf,
                                                     const float* relk,
                                                     const float* xmask) {
  const int t  = blockIdx.x;
  const int z  = blockIdx.y;       // b*H + h
  const int b  = z / H_;
  const int h  = z % H_;
  const int tid = threadIdx.x;
  __shared__ float qrow[DK_];
  __shared__ float red[256];
  const long long rowoff = ((long long)z * T_ + t) * T_;
  if (tid < DK_)
    qrow[tid] = (float)qbuf[((long long)b * T_ + t) * C_ + h * DK_ + tid];
  __syncthreads();
  const float mt = xmask[(long long)b * T_ + t];
  float v[4];
#pragma unroll
  for (int i = 0; i < 4; ++i) {
    const int s = tid + i * 256;
    float sc = scores[rowoff + s];
    const int e = s - t + W_;
    if (e >= 0 && e <= 2 * W_) {           // inside the 9-wide rel window
      const float* rk = relk + e * DK_;
      float d = 0.0f;
      for (int dd = 0; dd < DK_; ++dd) d += qrow[dd] * rk[dd];
      sc += d;
    }
    const float ms = xmask[(long long)b * T_ + s];
    v[i] = (mt == 0.0f || ms == 0.0f) ? -1e4f : sc;
  }
  float mx = fmaxf(fmaxf(v[0], v[1]), fmaxf(v[2], v[3]));
  red[tid] = mx; __syncthreads();
  for (int w = 128; w > 0; w >>= 1) { if (tid < w) red[tid] = fmaxf(red[tid], red[tid + w]); __syncthreads(); }
  mx = red[0]; __syncthreads();
  float sum = 0.0f;
#pragma unroll
  for (int i = 0; i < 4; ++i) { v[i] = __expf(v[i] - mx); sum += v[i]; }
  red[tid] = sum; __syncthreads();
  for (int w = 128; w > 0; w >>= 1) { if (tid < w) red[tid] += red[tid + w]; __syncthreads(); }
  const float inv = 1.0f / red[0];
#pragma unroll
  for (int i = 0; i < 4; ++i)
    P[rowoff + tid + i * 256] = (bf16_t)(v[i] * inv);
}

// out[b,t,h*64+d] += sum_e P[b,h,t,t+e-W] * rel_v[e,d]   (windowed rel-V term)
__global__ __launch_bounds__(64) void k_relv_add(const bf16_t* P, bf16_t* attn,
                                                 const float* relv) {
  const int t = blockIdx.x;
  const int z = blockIdx.y;
  const int b = z / H_, h = z % H_;
  const int d = threadIdx.x;
  const bf16_t* Prow = P + ((long long)z * T_ + t) * T_;
  float acc = 0.0f;
#pragma unroll
  for (int e = 0; e <= 2 * W_; ++e) {
    const int s = t + e - W_;
    if (s >= 0 && s < T_) acc += (float)Prow[s] * relv[e * DK_ + d];
  }
  bf16_t* o = attn + ((long long)b * T_ + t) * C_ + h * DK_ + d;
  *o = (bf16_t)((float)*o + acc);
}

// x32 = LN(x32 + y) over C; xbf = bf16(x32 * mask). One block per (t,b).
__global__ __launch_bounds__(256) void k_residual_ln(float* x32, bf16_t* xbf,
                                                     const float* y,
                                                     const float* gamma,
                                                     const float* beta,
                                                     const float* xmask) {
  const int t = blockIdx.x, b = blockIdx.y;
  const long long base = ((long long)b * T_ + t) * C_;
  __shared__ float red[256];
  const int tid = threadIdx.x;
  const float v0 = x32[base + tid]       + y[base + tid];
  const float v1 = x32[base + tid + 256] + y[base + tid + 256];
  red[tid] = v0 + v1; __syncthreads();
  for (int w = 128; w > 0; w >>= 1) { if (tid < w) red[tid] += red[tid + w]; __syncthreads(); }
  const float mean = red[0] * (1.0f / C_);
  __syncthreads();
  const float d0 = v0 - mean, d1 = v1 - mean;
  red[tid] = d0 * d0 + d1 * d1; __syncthreads();
  for (int w = 128; w > 0; w >>= 1) { if (tid < w) red[tid] += red[tid + w]; __syncthreads(); }
  const float rstd = rsqrtf(red[0] * (1.0f / C_) + 1e-5f);
  const float mk = xmask[(long long)b * T_ + t];
  const float o0 = d0 * rstd * gamma[tid]       + beta[tid];
  const float o1 = d1 * rstd * gamma[tid + 256] + beta[tid + 256];
  x32[base + tid]       = o0;  xbf[base + tid]       = (bf16_t)(o0 * mk);
  x32[base + tid + 256] = o1;  xbf[base + tid + 256] = (bf16_t)(o1 * mk);
}

// [B,C,T] -> [B,T,C] with mask; fp32 + bf16 copies
__global__ void k_transpose_in(const float* x, const float* xmask,
                               float* x32, bf16_t* xbf) {
  const long long i = (long long)blockIdx.x * blockDim.x + threadIdx.x;
  const int t = (int)(i % T_);
  long long r = i / T_;
  const int c = (int)(r % C_);
  const int b = (int)(r / C_);
  const float v = x[i] * xmask[(long long)b * T_ + t];
  const long long o = ((long long)b * T_ + t) * C_ + c;
  x32[o] = v; xbf[o] = (bf16_t)v;
}

// [B,T,C] -> [B,C,T] with mask
__global__ void k_final_out(const float* x32, const float* xmask, float* out) {
  const long long i = (long long)blockIdx.x * blockDim.x + threadIdx.x;
  const int t = (int)(i % T_);
  long long r = i / T_;
  const int c = (int)(r % C_);
  const int b = (int)(r / C_);
  out[i] = x32[((long long)b * T_ + t) * C_ + c] * xmask[(long long)b * T_ + t];
}

__global__ void k_f32_to_bf16(const float* in, bf16_t* out, long long n) {
  for (long long i = (long long)blockIdx.x * blockDim.x + threadIdx.x;
       i < n; i += (long long)gridDim.x * blockDim.x)
    out[i] = (bf16_t)in[i];
}

// in [L,O,I,3] (fp32) -> out [L,3,O,I] (bf16): tap-major repack for conv-GEMM
__global__ void k_pack_conv3(const float* in, bf16_t* out, int O, int I) {
  const long long n = (long long)L_ * O * I * 3;
  for (long long i = (long long)blockIdx.x * blockDim.x + threadIdx.x;
       i < n; i += (long long)gridDim.x * blockDim.x) {
    const int tap = (int)(i % 3);
    long long r = i / 3;
    const int ii = (int)(r % I); r /= I;
    const int o  = (int)(r % O);
    const int l  = (int)(r / O);
    out[(((long long)l * 3 + tap) * O + o) * I + ii] = (bf16_t)in[i];
  }
}

extern "C" void kernel_launch(void* const* d_in, const int* in_sizes, int n_in,
                              void* d_out, int out_size, void* d_ws, size_t ws_size,
                              hipStream_t stream) {
  (void)in_sizes; (void)n_in; (void)out_size; (void)ws_size;
  const float* x     = (const float*)d_in[0];
  const float* xmask = (const float*)d_in[1];
  const float* Wq = (const float*)d_in[2];   const float* bq = (const float*)d_in[3];
  const float* Wk = (const float*)d_in[4];   const float* bk = (const float*)d_in[5];
  const float* Wv = (const float*)d_in[6];   const float* bv = (const float*)d_in[7];
  const float* Wo = (const float*)d_in[8];   const float* bo = (const float*)d_in[9];
  const float* rel_k = (const float*)d_in[10]; const float* rel_v = (const float*)d_in[11];
  const float* g1 = (const float*)d_in[12];  const float* be1 = (const float*)d_in[13];
  const float* g2 = (const float*)d_in[14];  const float* be2 = (const float*)d_in[15];
  const float* Wf1 = (const float*)d_in[16]; const float* bf1 = (const float*)d_in[17];
  const float* Wf2 = (const float*)d_in[18]; const float* bf2 = (const float*)d_in[19];
  float* out = (float*)d_out;

  // ---- workspace layout ----
  char* ws = (char*)d_ws;
  size_t off = 0;
  auto take = [&](size_t bytes) -> void* {
    void* p = ws + off; off += (bytes + 255) & ~(size_t)255; return p;
  };
  float*  x32   = (float*) take(sizeof(float)  * (size_t)B_ * T_ * C_);
  bf16_t* xbf   = (bf16_t*)take(sizeof(bf16_t) * (size_t)B_ * T_ * C_);
  bf16_t* qb    = (bf16_t*)take(sizeof(bf16_t) * (size_t)B_ * T_ * C_);
  bf16_t* kb    = (bf16_t*)take(sizeof(bf16_t) * (size_t)B_ * T_ * C_);
  bf16_t* vb    = (bf16_t*)take(sizeof(bf16_t) * (size_t)B_ * T_ * C_);
  bf16_t* attn  = (bf16_t*)take(sizeof(bf16_t) * (size_t)B_ * T_ * C_);
  float*  scores = (float*) take(sizeof(float)  * (size_t)B_ * H_ * T_ * T_);
  bf16_t* Pbf    = (bf16_t*)take(sizeof(bf16_t) * (size_t)B_ * H_ * T_ * T_);
  bf16_t* Wqb = (bf16_t*)take(sizeof(bf16_t) * (size_t)L_ * C_ * C_);
  bf16_t* Wkb = (bf16_t*)take(sizeof(bf16_t) * (size_t)L_ * C_ * C_);
  bf16_t* Wvb = (bf16_t*)take(sizeof(bf16_t) * (size_t)L_ * C_ * C_);
  bf16_t* Wob = (bf16_t*)take(sizeof(bf16_t) * (size_t)L_ * C_ * C_);
  bf16_t* W1p = (bf16_t*)take(sizeof(bf16_t) * (size_t)L_ * 3 * F_ * C_);
  bf16_t* W2p = (bf16_t*)take(sizeof(bf16_t) * (size_t)L_ * 3 * C_ * F_);
  // y32/y1 alias the score buffer: scores are dead once softmax has produced P,
  // and are rewritten only at the next layer's score GEMM.
  float*  y32 = scores;
  bf16_t* y1  = (bf16_t*)((char*)scores + sizeof(float) * (size_t)B_ * T_ * C_);

  // ---- weight repack (bf16) + input transpose ----
  const long long nW = (long long)L_ * C_ * C_;
  k_f32_to_bf16<<<4096, 256, 0, stream>>>(Wq, Wqb, nW);
  k_f32_to_bf16<<<4096, 256, 0, stream>>>(Wk, Wkb, nW);
  k_f32_to_bf16<<<4096, 256, 0, stream>>>(Wv, Wvb, nW);
  k_f32_to_bf16<<<4096, 256, 0, stream>>>(Wo, Wob, nW);
  k_pack_conv3<<<8192, 256, 0, stream>>>(Wf1, W1p, F_, C_);
  k_pack_conv3<<<8192, 256, 0, stream>>>(Wf2, W2p, C_, F_);
  k_transpose_in<<<(B_ * C_ * T_) / 256, 256, 0, stream>>>(x, xmask, x32, xbf);

  auto gemm = [&](const bf16_t* A, int lda, long long sAb, long long sAh,
                  const bf16_t* Bp, int ldb, long long sBb, long long sBh, bool transB,
                  float* oF, bf16_t* oH, int ldc, long long sDb, long long sDh,
                  int M, int N, int K, int Hdim,
                  const float* bias, float alpha, int relu, int transOut,
                  const float* maskN, long long sMaskB,
                  int ntap, long long tapStrideA, int tapShift0) {
    GemmParams p;
    p.A = A; p.B = Bp; p.outF = oF; p.outH = oH; p.bias = bias; p.maskN = maskN;
    p.M = M; p.N = N; p.K = K; p.lda = lda; p.ldb = ldb; p.ldc = ldc;
    p.sAb = sAb; p.sAh = sAh; p.sBb = sBb; p.sBh = sBh; p.sDb = sDb; p.sDh = sDh;
    p.sMaskB = sMaskB; p.Hdim = Hdim;
    p.ntap = ntap; p.tapStrideA = tapStrideA; p.tapShift0 = tapShift0;
    p.alpha = alpha; p.relu = relu; p.transOut = transOut;
    dim3 grid(N / TILE_N, M / TILE_M, B_ * Hdim);
    if (transB) k_gemm<true ><<<grid, 256, 0, stream>>>(p);
    else        k_gemm<false><<<grid, 256, 0, stream>>>(p);
  };

  const long long sTC = (long long)T_ * C_;
  const long long sTT = (long long)T_ * T_;
  const float scale = 0.125f; // DK^-0.5

  for (int l = 0; l < L_; ++l) {
    const bf16_t* Wq_l = Wqb + (long long)l * C_ * C_;
    const bf16_t* Wk_l = Wkb + (long long)l * C_ * C_;
    const bf16_t* Wv_l = Wvb + (long long)l * C_ * C_;
    const bf16_t* Wo_l = Wob + (long long)l * C_ * C_;

    // q,k,v projections -> [B,T,C] bf16 (q pre-scaled, matches qs = q*scale)
    gemm(Wq_l, C_, 0, 0, xbf, C_, sTC, 0, true, nullptr, qb, C_, sTC, 0,
         C_, T_, C_, 1, bq + l * C_, scale, 0, 1, nullptr, 0, 1, 0, 0);
    gemm(Wk_l, C_, 0, 0, xbf, C_, sTC, 0, true, nullptr, kb, C_, sTC, 0,
         C_, T_, C_, 1, bk + l * C_, 1.0f, 0, 1, nullptr, 0, 1, 0, 0);
    gemm(Wv_l, C_, 0, 0, xbf, C_, sTC, 0, true, nullptr, vb, C_, sTC, 0,
         C_, T_, C_, 1, bv + l * C_, 1.0f, 0, 1, nullptr, 0, 1, 0, 0);

    // scores = qs @ k^T  -> fp32 [B,H,T,T]
    gemm(qb, C_, sTC, DK_, kb, C_, sTC, DK_, true,
         scores, nullptr, T_, (long long)H_ * sTT, sTT,
         T_, T_, DK_, H_, nullptr, 1.0f, 0, 0, nullptr, 0, 1, 0, 0);

    // softmax(scores + windowed rel-K bias, masked) -> P bf16
    k_softmax_rel<<<dim3(T_, B_ * H_), 256, 0, stream>>>(
        scores, Pbf, qb, rel_k + (long long)l * (2 * W_ + 1) * DK_, xmask);

    // out = P @ V -> attn [B,T,C] bf16 (per-head 64-col slice)
    gemm(Pbf, T_, (long long)H_ * sTT, sTT, vb, C_, sTC, DK_, false,
         nullptr, attn, C_, sTC, DK_,
         T_, DK_, T_, H_, nullptr, 1.0f, 0, 0, nullptr, 0, 1, 0, 0);

    // windowed rel-V term added into attn
    k_relv_add<<<dim3(T_, B_ * H_), 64, 0, stream>>>(
        Pbf, attn, rel_v + (long long)l * (2 * W_ + 1) * DK_);

    // y = Wo @ attn -> y32 fp32 [B,T,C]
    gemm(Wo_l, C_, 0, 0, attn, C_, sTC, 0, true, y32, nullptr, C_, sTC, 0,
         C_, T_, C_, 1, bo + l * C_, 1.0f, 0, 1, nullptr, 0, 1, 0, 0);

    // x = LN(x + y)
    k_residual_ln<<<dim3(T_, B_), 256, 0, stream>>>(
        x32, xbf, y32, g1 + l * C_, be1 + l * C_, xmask);

    // FFN1: relu(conv3(x*mask)) * mask -> y1 bf16 [B,T,F]
    gemm(W1p + (long long)l * 3 * F_ * C_, C_, 0, 0, xbf, C_, sTC, 0, true,
         nullptr, y1, F_, (long long)T_ * F_, 0,
         F_, T_, C_, 1, bf1 + l * F_, 1.0f, 1, 1, xmask, T_,
         3, (long long)F_ * C_, -1);

    // FFN2: conv3(y1) * mask -> y32 fp32 [B,T,C]
    gemm(W2p + (long long)l * 3 * C_ * F_, F_, 0, 0, y1, F_, (long long)T_ * F_, 0, true,
         y32, nullptr, C_, sTC, 0,
         C_, T_, F_, 1, bf2 + l * C_, 1.0f, 0, 1, xmask, T_,
         3, (long long)C_ * F_, -1);

    // x = LN(x + y)
    k_residual_ln<<<dim3(T_, B_), 256, 0, stream>>>(
        x32, xbf, y32, g2 + l * C_, be2 + l * C_, xmask);
  }

  k_final_out<<<(B_ * C_ * T_) / 256, 256, 0, stream>>>(x32, xmask, out);
}
